// TensorTrainComplexLinear_4629974745467
// MI455X (gfx1250) — matrix operand order repaired and checked by reading, not compiled, not run on hardware
//
#include <hip/hip_runtime.h>

// ---------------------------------------------------------------------------
// TT complex linear:  out = [xr*Wr^T - xi*Wi^T | xr*Wi^T + xi*Wr^T] + bias
//   x: (8*4096, 1024) fp32,  W_real/W_imag: 512x512 from TT cores.
// Strategy: bf16x3 compensated split-GEMM on v_wmma_f32_16x16x32_bf16.
//   - 4-wave WG (stacked in M), wave: M=16 x four N=16 tiles.
//   - Complex subtraction folded into a sign-flipped xi operand (v_xor),
//     so only TWO f32 accumulators per tile (64 VGPRs total) -> <256 VGPRs.
//   - All addresses: SGPR base + 32-bit lane byte offset + imm (GVS form).
// ---------------------------------------------------------------------------

typedef __attribute__((ext_vector_type(16))) __bf16       v16bf;
typedef __attribute__((ext_vector_type(8)))  float        v8f;
typedef __attribute__((ext_vector_type(4)))  float        v4f;
typedef __attribute__((ext_vector_type(4)))  unsigned int v4u;

union BF16x16 {
  v16bf    v;
  unsigned u[8];
  v4u      q[2];
};

#define OUTF    512
#define INF     512
#define ROWLEN  1024   // 2*INF (real | imag halves of each x / out row)

__device__ __forceinline__ unsigned bf16_rne_bits(float x) {
  unsigned u = __float_as_uint(x);
  return (u + 0x7FFFu + ((u >> 16) & 1u)) >> 16;   // round-to-nearest-even
}

// Split two fp32 into packed bf16-hi pair and packed bf16-lo (residual) pair.
// hi = truncate(x) (error absorbed exactly by lo); residual lowers to
// v_fma_mix_f32_bf16, packing via v_perm_b32.
__device__ __forceinline__ void split_pair(float x0, float x1,
                                           unsigned& hp, unsigned& lp) {
  unsigned u0 = __float_as_uint(x0);
  unsigned u1 = __float_as_uint(x1);
  hp = __builtin_amdgcn_perm(u1, u0, 0x07060302u);   // {hi16(x1), hi16(x0)}
  float r0 = x0 - __uint_as_float(u0 & 0xFFFF0000u);
  float r1 = x1 - __uint_as_float(u1 & 0xFFFF0000u);
  lp = __builtin_amdgcn_perm(__float_as_uint(r1), __float_as_uint(r0),
                             0x07060302u);
}

__device__ __forceinline__ void split4v(v4f f, unsigned* hu, unsigned* lu) {
  split_pair(f[0], f[1], hu[0], lu[0]);
  split_pair(f[2], f[3], hu[1], lu[1]);
}

__device__ __forceinline__ v8f wmma_bf16(v16bf a, v16bf b, v8f c) {
  return __builtin_amdgcn_wmma_f32_16x16x32_bf16(false, a, false, b, (short)0, c,
                                                 false, false);
}

__device__ __forceinline__ v4u ldB(const char* base, unsigned off) {
  return *(const v4u*)(base + off);
}

// ---------------------------------------------------------------------------
// Kernel 1: contract TT cores -> W_real/W_imag (fp32), store bf16 hi/lo planes.
// W[o0*16+o1, i0*16+i1] = sum_r G1[0,o0,r,i0] * G2[r,o1,0,i1]
// ---------------------------------------------------------------------------
__global__ __launch_bounds__(256)
void tt_build_w(const float* __restrict__ G1R, const float* __restrict__ G2R,
                const float* __restrict__ G1I, const float* __restrict__ G2I,
                unsigned short* __restrict__ WrH, unsigned short* __restrict__ WrL,
                unsigned short* __restrict__ WiH, unsigned short* __restrict__ WiL) {
  const int idx = blockIdx.x * 256 + threadIdx.x;   // 0 .. 512*512-1
  const int o = idx >> 9, i = idx & 511;
  const int o0 = o >> 4, o1 = o & 15, i0 = i >> 4, i1 = i & 15;

  float wr = 0.f, wi = 0.f;
#pragma unroll
  for (int r = 0; r < 4; ++r) {
    const int g1 = (o0 * 4 + r) * 32 + i0;   // G1 (1,32,4,32)
    const int g2 = (r * 16 + o1) * 16 + i1;  // G2 (4,16,1,16)
    wr += G1R[g1] * G2R[g2];
    wi += G1I[g1] * G2I[g2];
  }
  {
    unsigned hb = bf16_rne_bits(wr);
    float hf = __uint_as_float(hb << 16);
    WrH[idx] = (unsigned short)hb;
    WrL[idx] = (unsigned short)bf16_rne_bits(wr - hf);
  }
  {
    unsigned hb = bf16_rne_bits(wi);
    float hf = __uint_as_float(hb << 16);
    WiH[idx] = (unsigned short)hb;
    WiL[idx] = (unsigned short)bf16_rne_bits(wi - hf);
  }
}

// ---------------------------------------------------------------------------
// Kernel 2: complex GEMM.
// WG = 128 thr = 4 waves stacked in M. WG tile: M=64, N=64.
// Wave: M=16 rows x four N=16 tiles; 2 accumulators/tile (real, imag).
// K loop steps 32; bf16x3 (hh+hl+lh) per product; -xi via sign-flip XOR.
// ---------------------------------------------------------------------------
__global__ __launch_bounds__(128)
void tt_cgemm(const float* __restrict__ x,
              const unsigned short* __restrict__ WrHp,
              const unsigned short* __restrict__ WrLp,
              const unsigned short* __restrict__ WiHp,
              const unsigned short* __restrict__ WiLp,
              const float* __restrict__ bias_r,
              const float* __restrict__ bias_i,
              float* __restrict__ out) {
  const int tid  = threadIdx.x;
  const int lane = tid & 31;
  const int wave = tid >> 5;           // 0..3 (M direction)
  const int l16  = lane & 15;
  const int hl   = lane >> 4;          // wave half (selects K sub-groups)

  const unsigned mbase = blockIdx.y * 64u + wave * 16u;
  const unsigned nb_wg = blockIdx.x * 64u;

  const char* xc  = (const char*)x;
  const char* wrh = (const char*)WrHp;
  const char* wrl = (const char*)WrLp;
  const char* wih = (const char*)WiHp;
  const char* wil = (const char*)WiLp;

  const unsigned koff = hl * 8u;       // lanes 0-15: K 0..7/16..23 ; 16-31: 8..15/24..31

  // 32-bit lane byte offsets (kb = 0); kb/t/group contributions are immediates.
  const unsigned xofs  = ((mbase + l16) * ROWLEN + koff) * 4u;      // into x (real)
  const unsigned wofs0 = ((nb_wg + l16) * INF + koff) * 2u;         // into each W plane

  v8f acc[4][2];                       // [n-tile][real, imag]
  {
    v8f z = {};
#pragma unroll
    for (int t = 0; t < 4; ++t) { acc[t][0] = z; acc[t][1] = z; }
  }

#pragma unroll 2
  for (unsigned kb = 0; kb < INF; kb += 32) {
    const unsigned xo = xofs + kb * 4u;        // bytes (real); imag = +2048
    const unsigned wo = wofs0 + kb * 2u;       // bytes, shared by all 4 planes

    // ---- load fp32 A, split into bf16 hi/lo (ISA A-operand order) ----
    BF16x16 arH, arL, aiH, aiL, niH, niL;      // ni* = -xi (sign-flipped)
    {
      v4f f0 = *(const v4f*)(xc + xo);
      v4f f1 = *(const v4f*)(xc + xo + 16);
      v4f f2 = *(const v4f*)(xc + xo + 64);
      v4f f3 = *(const v4f*)(xc + xo + 80);
      split4v(f0, arH.u + 0, arL.u + 0);
      split4v(f1, arH.u + 2, arL.u + 2);
      split4v(f2, arH.u + 4, arL.u + 4);
      split4v(f3, arH.u + 6, arL.u + 6);
    }
    {
      v4f f0 = *(const v4f*)(xc + xo + 2048);
      v4f f1 = *(const v4f*)(xc + xo + 2048 + 16);
      v4f f2 = *(const v4f*)(xc + xo + 2048 + 64);
      v4f f3 = *(const v4f*)(xc + xo + 2048 + 80);
      split4v(f0, aiH.u + 0, aiL.u + 0);
      split4v(f1, aiH.u + 2, aiL.u + 2);
      split4v(f2, aiH.u + 4, aiL.u + 4);
      split4v(f3, aiH.u + 6, aiL.u + 6);
    }
    // -xi: exact bf16 negation (packed sign-bit flip)
#pragma unroll
    for (int k = 0; k < 8; ++k) {
      niH.u[k] = aiH.u[k] ^ 0x80008000u;
      niL.u[k] = aiL.u[k] ^ 0x80008000u;
    }

#pragma unroll
    for (unsigned t = 0; t < 4; ++t) {
      const unsigned ti = t * 16384u;          // t*16 rows * 512 * 2B (immediate)

      BF16x16 brH, brL, biH, biL;
      brH.q[0] = ldB(wrh, wo + ti);
      brH.q[1] = ldB(wrh, wo + ti + 32);
      brL.q[0] = ldB(wrl, wo + ti);
      brL.q[1] = ldB(wrl, wo + ti + 32);
      biH.q[0] = ldB(wih, wo + ti);
      biH.q[1] = ldB(wih, wo + ti + 32);
      biL.q[0] = ldB(wil, wo + ti);
      biL.q[1] = ldB(wil, wo + ti + 32);

      // real += xr*Wr + (-xi)*Wi   (bf16x3 each: hh + hl + lh)
      acc[t][0] = wmma_bf16(arH.v, brH.v, acc[t][0]);
      acc[t][0] = wmma_bf16(arH.v, brL.v, acc[t][0]);
      acc[t][0] = wmma_bf16(arL.v, brH.v, acc[t][0]);
      acc[t][0] = wmma_bf16(niH.v, biH.v, acc[t][0]);
      acc[t][0] = wmma_bf16(niH.v, biL.v, acc[t][0]);
      acc[t][0] = wmma_bf16(niL.v, biH.v, acc[t][0]);
      // imag += xr*Wi + xi*Wr
      acc[t][1] = wmma_bf16(arH.v, biH.v, acc[t][1]);
      acc[t][1] = wmma_bf16(arH.v, biL.v, acc[t][1]);
      acc[t][1] = wmma_bf16(arL.v, biH.v, acc[t][1]);
      acc[t][1] = wmma_bf16(aiH.v, brH.v, acc[t][1]);
      acc[t][1] = wmma_bf16(aiH.v, brL.v, acc[t][1]);
      acc[t][1] = wmma_bf16(aiL.v, brH.v, acc[t][1]);
    }
  }

  // ---- epilogue: bias add, scatter per C/D layout ----
  // byte offset of (row = mbase + hl*8, col = nb_wg + l16); j/t/imag are imm.
  const unsigned obase = ((mbase + hl * 8u) * ROWLEN + nb_wg + l16) * 4u;
  char* oc = (char*)out;
  const unsigned bofs = (nb_wg + l16) * 4u;
  const char* brc = (const char*)bias_r;
  const char* bic = (const char*)bias_i;

#pragma unroll
  for (unsigned t = 0; t < 4; ++t) {
    const float br = *(const float*)(brc + bofs + t * 64u);
    const float bi = *(const float*)(bic + bofs + t * 64u);
#pragma unroll
    for (unsigned j = 0; j < 8; ++j) {
      const unsigned o = obase + j * 4096u + t * 64u;   // j rows, t n-tiles
      *(float*)(oc + o)         = acc[t][0][j] + br;    // real
      *(float*)(oc + o + 2048u) = acc[t][1][j] + bi;    // imag
    }
  }
}

// ---------------------------------------------------------------------------
extern "C" void kernel_launch(void* const* d_in, const int* in_sizes, int n_in,
                              void* d_out, int out_size, void* d_ws, size_t ws_size,
                              hipStream_t stream) {
  const float* x   = (const float*)d_in[0];
  const float* G1R = (const float*)d_in[1];
  const float* G2R = (const float*)d_in[2];
  const float* G1I = (const float*)d_in[3];
  const float* G2I = (const float*)d_in[4];
  const float* br  = (const float*)d_in[5];
  const float* bi  = (const float*)d_in[6];

  // workspace: four 512x512 bf16 planes (2 MB total)
  unsigned short* WrH = (unsigned short*)d_ws;
  unsigned short* WrL = WrH + OUTF * INF;
  unsigned short* WiH = WrL + OUTF * INF;
  unsigned short* WiL = WiH + OUTF * INF;

  tt_build_w<<<(OUTF * INF) / 256, 256, 0, stream>>>(G1R, G2R, G1I, G2I,
                                                     WrH, WrL, WiH, WiL);

  // grid.x = N blocks (8) innermost so WGs sharing an x stripe co-run (L2 reuse);
  // grid.y = M stripes: 32768 / 64 = 512.
  dim3 grid(OUTF / 64, 32768 / 64, 1);
  tt_cgemm<<<grid, 128, 0, stream>>>(x, WrH, WrL, WiH, WiL, br, bi, (float*)d_out);
}
